// GaussianUpsampleIGAModule_74191265071608
// MI455X (gfx1250) — compile-verified, matrix operand order checked by reading, not run on hardware
//
#include <hip/hip_runtime.h>

typedef __attribute__((ext_vector_type(16))) _Float16 v16h;
typedef __attribute__((ext_vector_type(8)))  _Float16 v8h;
typedef __attribute__((ext_vector_type(8)))  float    v8f;

#define B_     4
#define KP_    512
#define M_     8
#define CS_    384
#define CS2_   768
#define COUT_  384
#define KC_    4096              // KP_*M_
#define NROW_  (B_*KP_)          // 2048
#define NKC_   (B_*KC_)          // 16384
#define PHI_   1.6f
#define LAM_   1e-4f
#define ETA_   1.0f
#define BETA_  0.0f
#define ALPHA_ 1.0f
#define EPS_   1e-6f

// ---------------------------------------------------------------------------
// WMMA wave32 fragment helpers (CDNA5 ISA 7.12.2 layouts)
// 16-bit A 16x32: lanes 0-15 hold K {0..7,16..23}, lanes 16-31 hold K {8..15,24..31}
// ---------------------------------------------------------------------------
__device__ __forceinline__ int frag_k(int lane, int e) {
  return ((lane >> 4) << 3) + e + ((e >> 3) << 3);
}

// A fragment from row-major [rows x ld] f16 (global or LDS, 16B-aligned rows):
// two contiguous 16-byte vector loads per lane, merged with a no-op shuffle.
__device__ __forceinline__ v16h load_frag_a(const _Float16* p, int ld, int row0,
                                            int k0, int lane) {
  const _Float16* r = p + (size_t)(row0 + (lane & 15)) * (size_t)ld + k0 +
                      ((lane >> 4) << 3);
  v8h lo = *(const v8h*)r;
  v8h hi = *(const v8h*)(r + 16);
  return __builtin_shufflevector(lo, hi, 0, 1, 2, 3, 4, 5, 6, 7,
                                 8, 9, 10, 11, 12, 13, 14, 15);
}

// B fragment from pre-packed fragment-order buffer: 32 contiguous bytes/lane.
__device__ __forceinline__ v16h load_frag_bp(const _Float16* p, int tile, int lane) {
  return *(const v16h*)(p + ((size_t)tile << 9) + (lane << 4));
}

__device__ __forceinline__ float wred_sum(float v) {
#pragma unroll
  for (int o = 16; o > 0; o >>= 1) v += __shfl_xor(v, o, 32);
  return v;
}
__device__ __forceinline__ float wred_max(float v) {
#pragma unroll
  for (int o = 16; o > 0; o >>= 1) v = fmaxf(v, __shfl_xor(v, o, 32));
  return v;
}
__device__ __forceinline__ float siluf(float x) { return __fdividef(x, 1.f + __expf(-x)); }
__device__ __forceinline__ float sigmf(float x) { return __fdividef(1.f, 1.f + __expf(-x)); }

// ---------------------------------------------------------------------------
// f32 -> f16 conversion
// ---------------------------------------------------------------------------
__global__ __launch_bounds__(256) void k_cvt(const float* __restrict__ s,
                                             _Float16* __restrict__ d, int n) {
  int i = blockIdx.x * 256 + threadIdx.x;
  if (i < n) d[i] = (_Float16)s[i];
}

// ---------------------------------------------------------------------------
// Pack f32 row-major [nmat][K][N] into f16 WMMA-B fragment order:
// tile = tn*(K/32)+tk ; within tile: lane*16+e holds (k = tk*32+frag_k(lane,e),
// n = tn*16+(lane&15)).  Each lane's 16 values are contiguous (32B).
// ---------------------------------------------------------------------------
__global__ __launch_bounds__(256) void k_pack_b(const float* __restrict__ src,
                                                _Float16* __restrict__ dst,
                                                int K, int N, int nmat) {
  int i = blockIdx.x * 256 + threadIdx.x;
  int per = K * N;
  if (i >= per * nmat) return;
  int m = i / per;
  int r = i - m * per;
  int tile = r >> 9;
  int within = r & 511;
  int lane = within >> 4, e = within & 15;
  int kt = K >> 5;
  int tn = tile / kt, tk = tile - tn * kt;
  int k = tk * 32 + ((lane >> 4) << 3) + e + ((e >> 3) << 3);
  int n = tn * 16 + (lane & 15);
  dst[i] = (_Float16)src[(size_t)m * per + (size_t)k * N + n];
}

// ---------------------------------------------------------------------------
// Kernel 1: prior MLP.  rows = B*KP.  a = softplus(silu(LN(s_l)@pW1+pb1)@pW2+pb2)*mask
// ---------------------------------------------------------------------------
__global__ __launch_bounds__(256) void k_prior(
    const float* __restrict__ s_l, const float* __restrict__ mask_l,
    const float* __restrict__ ln1_g, const float* __restrict__ ln1_b,
    const _Float16* __restrict__ pW1p, const float* __restrict__ pb1,
    const float* __restrict__ pW2, const float* __restrict__ pb2,
    float* __restrict__ a_out) {
  __shared__ __align__(32) _Float16 Aln[16][CS_];
  __shared__ float redA[16][16], redB[16][16], mstat[16][2], urow[16];
  const int t = threadIdx.x, lane = t & 31, wv = t >> 5;
  const int row0 = blockIdx.x * 16;
  const int rr = t >> 4, cseg = t & 15;
  if (t < 16) urow[t] = 0.f;
  const float* src = s_l + (size_t)(row0 + rr) * CS_;
  float s1 = 0.f, s2 = 0.f;
  for (int j = 0; j < 24; ++j) {
    float v = src[cseg * 24 + j];
    s1 += v; s2 += v * v;
  }
  redA[rr][cseg] = s1; redB[rr][cseg] = s2;
  __syncthreads();
  if (cseg == 0) {
    float a = 0.f, b = 0.f;
    for (int j = 0; j < 16; ++j) { a += redA[rr][j]; b += redB[rr][j]; }
    float mean = a / CS_;
    float var  = b / CS_ - mean * mean;
    mstat[rr][0] = mean; mstat[rr][1] = rsqrtf(var + 1e-5f);
  }
  __syncthreads();
  {
    float mean = mstat[rr][0], rstd = mstat[rr][1];
    for (int j = 0; j < 24; ++j) {
      int c = cseg * 24 + j;
      Aln[rr][c] = (_Float16)(((src[c] - mean) * rstd) * ln1_g[c] + ln1_b[c]);
    }
  }
  __syncthreads();
  // GEMM: [16 x 384] @ [384 x 384], each wave -> 48 cols (3 n-tiles)
  const int Kt = CS_ / 32;
  const int tn0 = wv * 3;
  v8f acc[3];
#pragma unroll
  for (int j = 0; j < 3; ++j)
#pragma unroll
    for (int v = 0; v < 8; ++v) acc[j][v] = 0.f;
  for (int tk = 0; tk < Kt; ++tk) {
    v16h af = load_frag_a(&Aln[0][0], CS_, 0, tk * 32, lane);
#pragma unroll
    for (int j = 0; j < 3; ++j) {
      v16h bf = load_frag_bp(pW1p, (tn0 + j) * Kt + tk, lane);
      acc[j] = __builtin_amdgcn_wmma_f32_16x16x32_f16(false, af, false, bf,
                                                      (short)0, acc[j], false, false);
    }
  }
#pragma unroll
  for (int j = 0; j < 3; ++j) {
    int col = (tn0 + j) * 16 + (lane & 15);
    float b1 = pb1[col], w2 = pW2[col];
#pragma unroll
    for (int v = 0; v < 8; ++v) {
      float h = siluf(acc[j][v] + b1);
      atomicAdd(&urow[v + ((lane >> 4) << 3)], h * w2);
    }
  }
  __syncthreads();
  if (t < 16) {
    float u = urow[t] + pb2[0];
    float sp = (u > 20.f) ? u : log1pf(__expf(u));
    a_out[row0 + t] = sp * mask_l[row0 + t];
  }
}

// ---------------------------------------------------------------------------
// Kernel 2: 3x3 symmetrize + jitter, Cholesky, inverse, logdet, mu0 candidates
// ---------------------------------------------------------------------------
__global__ __launch_bounds__(256) void k_chol(
    const float* __restrict__ Sigma_p, const float* __restrict__ mu_p,
    const float* __restrict__ xi,
    float* __restrict__ sig6, float* __restrict__ sinv6,
    float* __restrict__ logdet, float* __restrict__ mu0) {
  int idx = blockIdx.x * 256 + threadIdx.x;
  if (idx >= NROW_) return;
  const float* S = Sigma_p + (size_t)idx * 9;
  const float LJ = LAM_ + EPS_;
  float a00 = S[0] + LJ;
  float a01 = 0.5f * (S[1] + S[3]);
  float a02 = 0.5f * (S[2] + S[6]);
  float a11 = S[4] + LJ;
  float a12 = 0.5f * (S[5] + S[7]);
  float a22 = S[8] + LJ;
  float* sg = sig6 + (size_t)idx * 6;
  sg[0] = a00; sg[1] = a01; sg[2] = a02; sg[3] = a11; sg[4] = a12; sg[5] = a22;
  float l00 = sqrtf(a00);
  float l10 = a01 / l00, l20 = a02 / l00;
  float l11 = sqrtf(fmaxf(a11 - l10 * l10, 1e-12f));
  float l21 = (a12 - l20 * l10) / l11;
  float l22 = sqrtf(fmaxf(a22 - l20 * l20 - l21 * l21, 1e-12f));
  logdet[idx] = 2.f * (__logf(l00) + __logf(l11) + __logf(l22));
  float m00 = 1.f / l00, m11 = 1.f / l11, m22 = 1.f / l22;
  float m10 = -m11 * (l10 * m00);
  float m21 = -m22 * (l21 * m11);
  float m20 = -m22 * (l20 * m00 + l21 * m10);
  float* si = sinv6 + (size_t)idx * 6;
  si[0] = m00 * m00 + m10 * m10 + m20 * m20;
  si[1] = m10 * m11 + m20 * m21;
  si[2] = m20 * m22;
  si[3] = m11 * m11 + m21 * m21;
  si[4] = m21 * m22;
  si[5] = m22 * m22;
  const float* mp = mu_p + (size_t)idx * 3;
  const float* xr = xi + (size_t)idx * (M_ * 3);
  float* mo = mu0 + (size_t)idx * (M_ * 3);
#pragma unroll
  for (int m = 0; m < M_; ++m) {
    float x0 = xr[m * 3 + 0], x1 = xr[m * 3 + 1], x2 = xr[m * 3 + 2];
    mo[m * 3 + 0] = mp[0] + l00 * x0;
    mo[m * 3 + 1] = mp[1] + l10 * x0 + l11 * x1;
    mo[m * 3 + 2] = mp[2] + l20 * x0 + l21 * x1 + l22 * x2;
  }
}

// ---------------------------------------------------------------------------
// Kernel 3: gate MLP.  rows = B*Kc, K = 768 (concat LN), -> scalar gate g_flat
// ---------------------------------------------------------------------------
__global__ __launch_bounds__(256) void k_gate(
    const float* __restrict__ s_l, const float* __restrict__ mask_l,
    const float* __restrict__ e_embed,
    const float* __restrict__ ln2_g, const float* __restrict__ ln2_b,
    const _Float16* __restrict__ gW1p, const float* __restrict__ gb1,
    const float* __restrict__ gW2, const float* __restrict__ gb2,
    const float* __restrict__ a_in, float* __restrict__ gflat) {
  __shared__ __align__(32) _Float16 Aln[16][CS2_];
  __shared__ float redA[16][16], redB[16][16], mstat[16][2], urow[16];
  const int t = threadIdx.x, lane = t & 31, wv = t >> 5;
  const int row0 = blockIdx.x * 16;
  const int rr = t >> 4, cseg = t & 15;
  if (t < 16) urow[t] = 0.f;
  const int ri = row0 + rr;
  const int b  = ri >> 12;               // / KC_
  const int kp = (ri & (KC_ - 1)) >> 3;
  const int m  = ri & 7;
  const float* srow = s_l + (size_t)(b * KP_ + kp) * CS_;
  const float* erow = e_embed + (size_t)m * CS_;
  float s1 = 0.f, s2 = 0.f;
  for (int j = 0; j < 48; ++j) {
    int c = cseg * 48 + j;
    float v = (c < CS_) ? srow[c] : erow[c - CS_];
    s1 += v; s2 += v * v;
  }
  redA[rr][cseg] = s1; redB[rr][cseg] = s2;
  __syncthreads();
  if (cseg == 0) {
    float a = 0.f, bb = 0.f;
    for (int j = 0; j < 16; ++j) { a += redA[rr][j]; bb += redB[rr][j]; }
    float mean = a / CS2_;
    float var  = bb / CS2_ - mean * mean;
    mstat[rr][0] = mean; mstat[rr][1] = rsqrtf(var + 1e-5f);
  }
  __syncthreads();
  {
    float mean = mstat[rr][0], rstd = mstat[rr][1];
    for (int j = 0; j < 48; ++j) {
      int c = cseg * 48 + j;
      float v = (c < CS_) ? srow[c] : erow[c - CS_];
      Aln[rr][c] = (_Float16)(((v - mean) * rstd) * ln2_g[c] + ln2_b[c]);
    }
  }
  __syncthreads();
  // GEMM: [16 x 768] @ [768 x 384]
  const int Kt = CS2_ / 32;
  const int tn0 = wv * 3;
  v8f acc[3];
#pragma unroll
  for (int j = 0; j < 3; ++j)
#pragma unroll
    for (int v = 0; v < 8; ++v) acc[j][v] = 0.f;
  for (int tk = 0; tk < Kt; ++tk) {
    v16h af = load_frag_a(&Aln[0][0], CS2_, 0, tk * 32, lane);
#pragma unroll
    for (int j = 0; j < 3; ++j) {
      v16h bf = load_frag_bp(gW1p, (tn0 + j) * Kt + tk, lane);
      acc[j] = __builtin_amdgcn_wmma_f32_16x16x32_f16(false, af, false, bf,
                                                      (short)0, acc[j], false, false);
    }
  }
#pragma unroll
  for (int j = 0; j < 3; ++j) {
    int col = (tn0 + j) * 16 + (lane & 15);
    float b1 = gb1[col], w2 = gW2[col];
#pragma unroll
    for (int v = 0; v < 8; ++v) {
      float h = siluf(acc[j][v] + b1);
      atomicAdd(&urow[v + ((lane >> 4) << 3)], h * w2);
    }
  }
  __syncthreads();
  if (t < 16) {
    int rig = row0 + t;
    int bb  = rig >> 12;
    int kpp = (rig & (KC_ - 1)) >> 3;
    float mk = mask_l[bb * KP_ + kpp];
    float blog = urow[t] + gb2[0] + BETA_ * __logf(a_in[bb * KP_ + kpp] + EPS_);
    float g = sigmf(blog / ETA_) * mk;
    gflat[rig] = g * mk;   // g * mask0
  }
}

// ---------------------------------------------------------------------------
// Kernel 4: responsibilities (one wave per child row) + moment matching + gf
// ---------------------------------------------------------------------------
__global__ __launch_bounds__(256) void k_resp(
    const float* __restrict__ mu_p, const float* __restrict__ mask_l,
    const float* __restrict__ sig6, const float* __restrict__ sinv6,
    const float* __restrict__ logdet, const float* __restrict__ mu0,
    _Float16* __restrict__ bresp, float* __restrict__ gfeat) {
  const int lane = threadIdx.x & 31;
  const int ri = blockIdx.x * 8 + (threadIdx.x >> 5);
  const int b = ri >> 12;
  const float* m0 = mu0 + (size_t)ri * 3;
  const float x0 = m0[0], x1 = m0[1], x2 = m0[2];
  float sc[16];
  float mx = -3.4e38f;
#pragma unroll
  for (int j = 0; j < 16; ++j) {
    int k = b * KP_ + lane + (j << 5);
    const float* si = sinv6 + (size_t)k * 6;
    const float* mp = mu_p + (size_t)k * 3;
    float d0 = x0 - mp[0], d1 = x1 - mp[1], d2 = x2 - mp[2];
    float maha = si[0]*d0*d0 + si[3]*d1*d1 + si[5]*d2*d2
               + 2.f*(si[1]*d0*d1 + si[2]*d0*d2 + si[4]*d1*d2);
    float s = -0.5f * maha - 0.5f * logdet[k];
    if (mask_l[k] < 0.5f) s = -1e9f;
    sc[j] = s;
    mx = fmaxf(mx, s);
  }
  mx = wred_max(mx);
  float sum = 0.f;
#pragma unroll
  for (int j = 0; j < 16; ++j) { sc[j] = __expf(ALPHA_ * (sc[j] - mx)); sum += sc[j]; }
  sum = wred_sum(sum);
  const float inv = __fdividef(1.f, sum);
  const float ip2 = 1.f / (PHI_ * PHI_);
  float mc0=0,mc1=0,mc2=0;
  float q00=0,q01=0,q02=0,q11=0,q12=0,q22=0;     // sum p * mu mu^T
  float i00=0,i01=0,i02=0,i11=0,i12=0,i22=0;     // sum p * Sig_j
#pragma unroll
  for (int j = 0; j < 16; ++j) {
    int kk = lane + (j << 5);
    int k  = b * KP_ + kk;
    float p = sc[j] * inv;
    const float* mp = mu_p + (size_t)k * 3;
    float p0 = mp[0], p1 = mp[1], p2 = mp[2];
    mc0 += p*p0; mc1 += p*p1; mc2 += p*p2;
    q00 += p*p0*p0; q01 += p*p0*p1; q02 += p*p0*p2;
    q11 += p*p1*p1; q12 += p*p1*p2; q22 += p*p2*p2;
    const float* sg = sig6 + (size_t)k * 6;
    i00 += p*sg[0]; i01 += p*sg[1]; i02 += p*sg[2];
    i11 += p*sg[3]; i12 += p*sg[4]; i22 += p*sg[5];
    bresp[(size_t)ri * KP_ + kk] = (_Float16)p;
  }
  mc0 = wred_sum(mc0); mc1 = wred_sum(mc1); mc2 = wred_sum(mc2);
  q00 = wred_sum(q00); q01 = wred_sum(q01); q02 = wred_sum(q02);
  q11 = wred_sum(q11); q12 = wred_sum(q12); q22 = wred_sum(q22);
  i00 = wred_sum(i00); i01 = wred_sum(i01); i02 = wred_sum(i02);
  i11 = wred_sum(i11); i12 = wred_sum(i12); i22 = wred_sum(i22);
  if (lane == 0) {
    float s00 = i00*ip2 + q00 - mc0*mc0 + LAM_;
    float s01 = i01*ip2 + q01 - mc0*mc1;
    float s02 = i02*ip2 + q02 - mc0*mc2;
    float s11 = i11*ip2 + q11 - mc1*mc1 + LAM_;
    float s12 = i12*ip2 + q12 - mc1*mc2;
    float s22 = i22*ip2 + q22 - mc2*mc2 + LAM_;
    float det = s00*(s11*s22 - s12*s12) - s01*(s01*s22 - s02*s12)
              + s02*(s01*s12 - s02*s11);
    float d0 = fmaxf(s00, 1e-8f), d1 = fmaxf(s11, 1e-8f), d2 = fmaxf(s22, 1e-8f);
    float* gf = gfeat + (size_t)ri * 8;
    gf[0] = mc0; gf[1] = mc1; gf[2] = mc2;
    gf[3] = __logf(fmaxf(det, 1e-8f));
    gf[4] = __logf(d0 + d1 + d2 + 1e-8f);
    gf[5] = __logf(d0); gf[6] = __logf(d1); gf[7] = __logf(d2);
  }
}

// ---------------------------------------------------------------------------
// Kernel 5: t = B_resp @ s_l  (per batch: 4096x512 @ 512x384), f16 out
// ---------------------------------------------------------------------------
__global__ __launch_bounds__(256) void k_bs(
    const _Float16* __restrict__ bresp, const _Float16* __restrict__ slp,
    _Float16* __restrict__ t16) {
  const int lane = threadIdx.x & 31, wv = threadIdx.x >> 5;
  const int row0 = blockIdx.x * 16;
  const int b = row0 >> 12;
  const _Float16* Bp = slp + (size_t)b * KP_ * CS_;   // packed per batch
  const int Kt = KP_ / 32;                            // 16
  const int tn0 = wv * 3;
  v8f acc[3];
#pragma unroll
  for (int j = 0; j < 3; ++j)
#pragma unroll
    for (int v = 0; v < 8; ++v) acc[j][v] = 0.f;
  for (int tk = 0; tk < Kt; ++tk) {
    v16h af = load_frag_a(bresp, KP_, row0, tk * 32, lane);
    if (tk + 1 < Kt)
      __builtin_prefetch(bresp + (size_t)(row0 + (lane & 15)) * KP_ + (tk + 1) * 32, 0, 0);
#pragma unroll
    for (int j = 0; j < 3; ++j) {
      v16h bf = load_frag_bp(Bp, (tn0 + j) * Kt + tk, lane);
      acc[j] = __builtin_amdgcn_wmma_f32_16x16x32_f16(false, af, false, bf,
                                                      (short)0, acc[j], false, false);
    }
  }
#pragma unroll
  for (int j = 0; j < 3; ++j) {
    int col = (tn0 + j) * 16 + (lane & 15);
#pragma unroll
    for (int v = 0; v < 8; ++v) {
      int rg = row0 + v + ((lane >> 4) << 3);
      t16[(size_t)rg * CS_ + col] = (_Float16)acc[j][v];
    }
  }
}

// ---------------------------------------------------------------------------
// Kernel 6: out = (t@semW + silu(gf@geoW1+geob1)@geoW2 + semb + geob2) * mask
// Both K=384 GEMMs fused into the same accumulators.
// ---------------------------------------------------------------------------
__global__ __launch_bounds__(256) void k_out(
    const _Float16* __restrict__ t16, const float* __restrict__ gfeat,
    const float* __restrict__ gflat,
    const _Float16* __restrict__ semWp, const float* __restrict__ semb,
    const float* __restrict__ geoW1, const float* __restrict__ geob1,
    const _Float16* __restrict__ geoW2p, const float* __restrict__ geob2,
    float* __restrict__ out) {
  __shared__ __align__(32) _Float16 hid[16][COUT_];
  __shared__ float gfs[16][8];
  const int t = threadIdx.x, lane = t & 31, wv = t >> 5;
  const int row0 = blockIdx.x * 16;
  if (t < 128) gfs[t >> 3][t & 7] = gfeat[(size_t)(row0 + (t >> 3)) * 8 + (t & 7)];
  __syncthreads();
  for (int e = t; e < 16 * COUT_; e += 256) {
    int r = e / COUT_, n = e % COUT_;
    float h = geob1[n];
#pragma unroll
    for (int j = 0; j < 8; ++j) h += gfs[r][j] * geoW1[j * COUT_ + n];
    hid[r][n] = (_Float16)siluf(h);
  }
  __syncthreads();
  const int Kt = CS_ / 32;
  const int tn0 = wv * 3;
  v8f acc[3];
#pragma unroll
  for (int j = 0; j < 3; ++j)
#pragma unroll
    for (int v = 0; v < 8; ++v) acc[j][v] = 0.f;
  for (int tk = 0; tk < Kt; ++tk) {
    v16h a1 = load_frag_a(t16, CS_, row0, tk * 32, lane);
    v16h a2 = load_frag_a(&hid[0][0], COUT_, 0, tk * 32, lane);
    if (tk + 1 < Kt)
      __builtin_prefetch(t16 + (size_t)(row0 + (lane & 15)) * CS_ + (tk + 1) * 32, 0, 0);
#pragma unroll
    for (int j = 0; j < 3; ++j) {
      v16h b1 = load_frag_bp(semWp, (tn0 + j) * Kt + tk, lane);
      acc[j] = __builtin_amdgcn_wmma_f32_16x16x32_f16(false, a1, false, b1,
                                                      (short)0, acc[j], false, false);
      v16h b2 = load_frag_bp(geoW2p, (tn0 + j) * Kt + tk, lane);
      acc[j] = __builtin_amdgcn_wmma_f32_16x16x32_f16(false, a2, false, b2,
                                                      (short)0, acc[j], false, false);
    }
  }
#pragma unroll
  for (int j = 0; j < 3; ++j) {
    int col = (tn0 + j) * 16 + (lane & 15);
    float bias = semb[col] + geob2[col];
#pragma unroll
    for (int v = 0; v < 8; ++v) {
      int rg = row0 + v + ((lane >> 4) << 3);
      float mc = fminf(fmaxf(gflat[rg], 0.f), 1.f);
      out[(size_t)rg * COUT_ + col] = (acc[j][v] + bias) * mc;
    }
  }
}

// ---------------------------------------------------------------------------
extern "C" void kernel_launch(void* const* d_in, const int* in_sizes, int n_in,
                              void* d_out, int out_size, void* d_ws, size_t ws_size,
                              hipStream_t stream) {
  const float* s_l     = (const float*)d_in[0];
  const float* mask_l  = (const float*)d_in[1];
  const float* mu_p    = (const float*)d_in[2];
  const float* Sigma_p = (const float*)d_in[3];
  const float* xi      = (const float*)d_in[4];
  const float* ln1_g   = (const float*)d_in[5];
  const float* ln1_b   = (const float*)d_in[6];
  const float* pW1     = (const float*)d_in[7];
  const float* pb1     = (const float*)d_in[8];
  const float* pW2     = (const float*)d_in[9];
  const float* pb2     = (const float*)d_in[10];
  const float* e_embed = (const float*)d_in[11];
  const float* ln2_g   = (const float*)d_in[12];
  const float* ln2_b   = (const float*)d_in[13];
  const float* gW1     = (const float*)d_in[14];
  const float* gb1     = (const float*)d_in[15];
  const float* gW2     = (const float*)d_in[16];
  const float* gb2     = (const float*)d_in[17];
  const float* semW    = (const float*)d_in[18];
  const float* semb    = (const float*)d_in[19];
  const float* geoW1   = (const float*)d_in[20];
  const float* geob1   = (const float*)d_in[21];
  const float* geoW2   = (const float*)d_in[22];
  const float* geob2   = (const float*)d_in[23];
  float* out = (float*)d_out;
  (void)in_sizes; (void)n_in; (void)out_size; (void)ws_size;

  char* ws = (char*)d_ws;
  size_t off = 0;
  auto carve = [&](size_t bytes) {
    void* p = ws + off;
    off = (off + bytes + 255) & ~(size_t)255;
    return p;
  };
  _Float16* slp    = (_Float16*)carve((size_t)NROW_ * CS_ * 2);   // packed, per batch
  _Float16* pW1p   = (_Float16*)carve((size_t)CS_ * CS_ * 2);
  _Float16* gW1p   = (_Float16*)carve((size_t)CS2_ * CS_ * 2);
  _Float16* semWp  = (_Float16*)carve((size_t)CS_ * COUT_ * 2);
  _Float16* geoW2p = (_Float16*)carve((size_t)COUT_ * COUT_ * 2);
  _Float16* bresp  = (_Float16*)carve((size_t)NKC_ * KP_ * 2);
  _Float16* t16    = (_Float16*)carve((size_t)NKC_ * CS_ * 2);
  float* a_ws   = (float*)carve((size_t)NROW_ * 4);
  float* sig6   = (float*)carve((size_t)NROW_ * 6 * 4);
  float* sinv6  = (float*)carve((size_t)NROW_ * 6 * 4);
  float* logdet = (float*)carve((size_t)NROW_ * 4);
  float* mu0    = (float*)carve((size_t)NKC_ * 3 * 4);
  float* gfeat  = (float*)carve((size_t)NKC_ * 8 * 4);
  float* gflat  = (float*)carve((size_t)NKC_ * 4);

  auto packb = [&](const float* s, _Float16* d, int K, int N, int nmat) {
    int n = K * N * nmat;
    k_pack_b<<<(n + 255) / 256, 256, 0, stream>>>(s, d, K, N, nmat);
  };
  packb(s_l, slp, KP_, CS_, B_);          // s_l as B-operand, per batch
  packb(pW1, pW1p, CS_, CS_, 1);
  packb(gW1, gW1p, CS2_, CS_, 1);
  packb(semW, semWp, CS_, COUT_, 1);
  packb(geoW2, geoW2p, COUT_, COUT_, 1);

  k_prior<<<NROW_ / 16, 256, 0, stream>>>(s_l, mask_l, ln1_g, ln1_b, pW1p, pb1,
                                          pW2, pb2, a_ws);
  k_chol<<<(NROW_ + 255) / 256, 256, 0, stream>>>(Sigma_p, mu_p, xi, sig6, sinv6,
                                                  logdet, mu0);
  k_gate<<<NKC_ / 16, 256, 0, stream>>>(s_l, mask_l, e_embed, ln2_g, ln2_b, gW1p,
                                        gb1, gW2, gb2, a_ws, gflat);
  k_resp<<<NKC_ / 8, 256, 0, stream>>>(mu_p, mask_l, sig6, sinv6, logdet, mu0,
                                       bresp, gfeat);
  k_bs<<<NKC_ / 16, 256, 0, stream>>>(bresp, slp, t16);
  k_out<<<NKC_ / 16, 256, 0, stream>>>(t16, gfeat, gflat, semWp, semb, geoW1,
                                       geob1, geoW2p, geob2, out);
}
